// RankMixerBlock_42743514530359
// MI455X (gfx1250) — matrix operand order repaired and checked by reading, not compiled, not run on hardware
//
#include <hip/hip_runtime.h>
#include <hip/hip_bf16.h>

// ---- problem constants (B,T,D,E,FH from reference) ----
#define BN  512
#define TN  16
#define DN  512
#define EN  4
#define FHN 2048

typedef __attribute__((ext_vector_type(16))) _Float16     v16h;
typedef __attribute__((ext_vector_type(2)))  _Float16     h2;
typedef __attribute__((ext_vector_type(8)))  float        v8f;
typedef __attribute__((ext_vector_type(8)))  unsigned int v8u;
typedef __attribute__((ext_vector_type(4)))  unsigned int u32x4;
typedef __attribute__((ext_vector_type(8)))  int          i32x8;
typedef __attribute__((ext_vector_type(4)))  int          i32x4;

__device__ __forceinline__ unsigned short f2h(float f) {
  return __builtin_bit_cast(unsigned short, (_Float16)f);
}
__device__ __forceinline__ unsigned pack2(float a, float b) {
  h2 h; h.x = (_Float16)a; h.y = (_Float16)b;      // -> v_cvt_pk_f16_f32
  return __builtin_bit_cast(unsigned, h);
}
__device__ __forceinline__ v16h frag_from(uint4 a, uint4 b) {
  v8u u;
  u[0] = a.x; u[1] = a.y; u[2] = a.z; u[3] = a.w;
  u[4] = b.x; u[5] = b.y; u[6] = b.z; u[7] = b.w;
  return __builtin_bit_cast(v16h, u);
}

// ---- Tensor Data Mover: 2D tile (32 halfs x 128 rows) global -> LDS with
//      LDS pad (16 dwords data + 4 dwords pad = 40-half pitch). ISA ch.8 D#.
//      6-arg builtin form (clang-23 / therock headers).
__device__ __forceinline__ void tdm_load_a_tile(const unsigned short* gsrc,
                                                int lda, unsigned ldsByteAddr)
{
  unsigned long long ga = (unsigned long long)(size_t)(const void*)gsrc;
  u32x4 g0;
  g0[0] = 1u;                                             // count=1 (valid, user)
  g0[1] = ldsByteAddr;                                    // lds_addr
  g0[2] = (unsigned)(ga & 0xffffffffull);                 // global_addr[31:0]
  g0[3] = (unsigned)((ga >> 32) & 0x1ffffffull) | (2u << 30); // addr[56:32]|type=2
  i32x8 g1;
  g1[0] = (1 << 16)        // data_size = 2 bytes
        | (1 << 20)        // pad_enable
        | (3 << 22)        // pad_interval = 16 dwords (one 32-half row)
        | (3 << 25);       // pad_amount   = 4 dwords (8 halfs) -> pitch 40
  g1[1] = 0;                           // atomic barrier addr=0; tensor_dim0 lo16=0
  g1[2] = 0x4000;                      // tensor_dim0 = 1<<30 (no OOB)
  g1[3] = 0x4000 | (32 << 16);         // tensor_dim1 = 1<<30; tile_dim0 = 32
  g1[4] = 128;                         // tile_dim1 = 128 rows; tile_dim2 = 0
  g1[5] = lda;                         // tensor_dim0_stride lo32 (elements)
  g1[6] = 0;
  g1[7] = 0;
  i32x4 gz4 = {0, 0, 0, 0};
  i32x8 gz8 = {0, 0, 0, 0, 0, 0, 0, 0};
  __builtin_amdgcn_tensor_load_to_lds(g0, g1, gz4, gz4, gz8, 0);
}

// =====================================================================
// WMMA GEMM: C[z] = A[z](MxK) * B[z](KxN).  AKIND: 0 = A fp32 (cvt in
// staging), 1 = A f16 (tile staged by the Tensor Data Mover).
// EPI: 0 = fp32 store, 1 = +bias, exact GELU, f16 store.
// 128x128x32 block tile, 256 threads = 8 waves (2x4), register-staged
// double-buffered LDS pipeline, LDS-transposed coalesced epilogue.
// =====================================================================
#define MBLK 128
#define NBLK 128
#define KBLK 32
#define APITCH 40     // halfs per A row in LDS
#define BPITCH 40     // halfs per B column in LDS
#define ATILE (MBLK * APITCH)   // 5120 halfs
#define BTILE (NBLK * BPITCH)   // 5120 halfs

template <int AKIND, int EPI>
__global__ __launch_bounds__(256) void gemm_wmma(
    const void* __restrict__ Agv, int lda, long long aStride, int aDiv,
    const float* __restrict__ Bg, int ldb, long long bStride,
    void* __restrict__ Cgv, int ldc, long long cStride,
    const float* __restrict__ biasBase, int biasStride, int Ksz)
{
  __shared__ __align__(16) unsigned short smem[2 * ATILE + 2 * BTILE];

  const int tid  = threadIdx.x;
  const int lane = tid & 31;
  const int wave = tid >> 5;
  const int ln   = lane & 15;
  const int hh   = lane >> 4;
  const int wm   = wave & 1;
  const int wn   = wave >> 1;

  const int z       = blockIdx.z;
  const int rowBase = blockIdx.y * MBLK;
  const int colBase = blockIdx.x * NBLK;
  const float* Bz   = Bg + (long long)z * bStride;

  // loop-invariant staging indices
  const int aR  = tid >> 3;                  // AKIND0: float4 id -> row, 8/row
  const int aKc = (tid & 7) << 2;
  const int bKk = tid >> 5;                  // B: float4 id -> k row, 32/row
  const int bNc = (tid & 31) << 2;

  v8f acc[4][2] = {};
  float4 aReg[4];
  float4 bReg[4];

  const int nk = Ksz / KBLK;

  // ---- prologue: fetch tile 0 ----
  #pragma unroll
  for (int c = 0; c < 4; ++c)
    bReg[c] = *(const float4*)(Bz + (long long)((c << 3) + bKk) * ldb + colBase + bNc);
  if constexpr (AKIND == 0) {
    const float* Ag = (const float*)Agv + (long long)(z / aDiv) * aStride;
    #pragma unroll
    for (int c = 0; c < 4; ++c)
      aReg[c] = *(const float4*)(Ag + (long long)(rowBase + aR + c * 32) * lda + aKc);
  } else {
    const unsigned short* Ag =
        (const unsigned short*)Agv + (long long)(z / aDiv) * aStride;
    if (wave == 0)
      tdm_load_a_tile(Ag + (long long)rowBase * lda,
                      lda, (unsigned)(size_t)(void*)smem);
  }

  for (int kt = 0; kt < nk; ++kt) {
    const int buf = kt & 1;
    unsigned short* Alds = smem + buf * ATILE;
    unsigned short* Blds = smem + 2 * ATILE + buf * BTILE;

    // ---- commit staged tile to LDS ----
    if constexpr (AKIND == 0) {
      #pragma unroll
      for (int c = 0; c < 4; ++c) {
        uint2 p;
        p.x = pack2(aReg[c].x, aReg[c].y);
        p.y = pack2(aReg[c].z, aReg[c].w);
        *(uint2*)&Alds[(aR + c * 32) * APITCH + aKc] = p;
      }
    }
    #pragma unroll
    for (int c = 0; c < 4; ++c) {
      // linear = tid + c*256 -> kk = bKk + c*8, nc = bNc
      int kk = bKk + (c << 3);
      int bh = kk >> 4, bj = kk & 15;
      Blds[(bNc + 0) * BPITCH + bh * 16 + bj] = f2h(bReg[c].x);
      Blds[(bNc + 1) * BPITCH + bh * 16 + bj] = f2h(bReg[c].y);
      Blds[(bNc + 2) * BPITCH + bh * 16 + bj] = f2h(bReg[c].z);
      Blds[(bNc + 3) * BPITCH + bh * 16 + bj] = f2h(bReg[c].w);
    }
    if constexpr (AKIND == 1) {
      if (wave == 0) __builtin_amdgcn_s_wait_tensorcnt((short)0);  // A tile landed
    }
    __syncthreads();

    // ---- fetch tile kt+1 (overlaps with WMMA below) ----
    if (kt + 1 < nk) {
      const int k0n = (kt + 1) * KBLK;
      #pragma unroll
      for (int c = 0; c < 4; ++c)
        bReg[c] = *(const float4*)(Bz + (long long)(k0n + bKk + (c << 3)) * ldb +
                                   colBase + bNc);
      if constexpr (AKIND == 0) {
        const float* Ag = (const float*)Agv + (long long)(z / aDiv) * aStride;
        #pragma unroll
        for (int c = 0; c < 4; ++c)
          aReg[c] = *(const float4*)(Ag + (long long)(rowBase + aR + c * 32) * lda +
                                     k0n + aKc);
      } else {
        const unsigned short* Ag =
            (const unsigned short*)Agv + (long long)(z / aDiv) * aStride;
        if (wave == 0)
          tdm_load_a_tile(Ag + (long long)rowBase * lda + k0n, lda,
                          (unsigned)(size_t)(void*)(smem + (buf ^ 1) * ATILE));
      }
      if (kt + 2 < nk) {      // deep prefetch of B (global_prefetch_b8)
        int kp = (kt + 2) * KBLK;
        __builtin_prefetch(
            (const void*)(Bz + (long long)(kp + bKk) * ldb + colBase + bNc), 0, 1);
      }
    }

    // ---- fragments + WMMA (ISA 7.12.2 layouts) ----
    v16h afrag[4], bfrag[2];
    #pragma unroll
    for (int mi = 0; mi < 4; ++mi) {
      int r = wm * 64 + mi * 16 + ln;
      uint4 x0 = *(const uint4*)&Alds[r * APITCH + hh * 8];
      uint4 x1 = *(const uint4*)&Alds[r * APITCH + 16 + hh * 8];
      afrag[mi] = frag_from(x0, x1);
    }
    #pragma unroll
    for (int ni = 0; ni < 2; ++ni) {
      int n = wn * 32 + ni * 16 + ln;
      uint4 x0 = *(const uint4*)&Blds[n * BPITCH + hh * 8];
      uint4 x1 = *(const uint4*)&Blds[n * BPITCH + 16 + hh * 8];
      bfrag[ni] = frag_from(x0, x1);
    }
    #pragma unroll
    for (int mi = 0; mi < 4; ++mi)
      #pragma unroll
      for (int ni = 0; ni < 2; ++ni)
        acc[mi][ni] = __builtin_amdgcn_wmma_f32_16x16x32_f16(
            false, afrag[mi], false, bfrag[ni], (short)0, acc[mi][ni], false, false);
  }

  // ---- epilogue: stage through LDS for coalesced global stores ----
  float*          epiF = (float*)smem;           // 64 x 132 fp32 (33.8 KB)
  unsigned short* epiH = (unsigned short*)smem;  // 64 x 136 f16  (17.4 KB)
  const long long cOff = (long long)z * cStride;
  #pragma unroll
  for (int mh = 0; mh < 2; ++mh) {
    __syncthreads();
    if (wm == mh) {
      #pragma unroll
      for (int mi = 0; mi < 4; ++mi) {
        #pragma unroll
        for (int ni = 0; ni < 2; ++ni) {
          int nloc = wn * 32 + ni * 16 + ln;
          #pragma unroll
          for (int r = 0; r < 8; ++r) {
            int lr = mi * 16 + hh * 8 + r;     // 0..63
            float val = acc[mi][ni][r];
            if constexpr (EPI == 0) {
              epiF[lr * 132 + nloc] = val;
            } else {
              val += biasBase[(long long)z * biasStride + colBase + nloc];
              float g = 0.5f * val * (1.0f + erff(val * 0.70710678118654752f));
              epiH[lr * 136 + nloc] = f2h(g);
            }
          }
        }
      }
    }
    __syncthreads();
    if constexpr (EPI == 0) {
      float* Cg = (float*)Cgv + cOff;
      #pragma unroll
      for (int c = 0; c < 8; ++c) {
        int id = tid + c * 256;                // float4 id, 32 per row
        int rr = id >> 5, c4 = (id & 31) << 2;
        float4 v = *(const float4*)&epiF[rr * 132 + c4];
        *(float4*)(Cg + (long long)(rowBase + mh * 64 + rr) * ldc + colBase + c4) = v;
      }
    } else {
      unsigned short* Cg = (unsigned short*)Cgv + cOff;
      #pragma unroll
      for (int c = 0; c < 4; ++c) {
        int id = tid + c * 256;                // uint4 id, 16 per row
        int rr = id >> 4, c8 = (id & 15) << 3;
        uint4 v = *(const uint4*)&epiH[rr * 136 + c8];
        *(uint4*)(Cg + (long long)(rowBase + mh * 64 + rr) * ldc + colBase + c8) = v;
      }
    }
  }
}

// =====================================================================
// Helper kernels
// =====================================================================
__global__ void zero_aux_kernel(float* aux) { *aux = 0.0f; }

// Amix[(b*16+h)*512 + t*32 + j] = x[(b*16+t)*512 + h*32 + j]
__global__ __launch_bounds__(256) void mix_kernel(
    const float* __restrict__ x, float* __restrict__ Amix)
{
  long long i = (long long)blockIdx.x * 256 + threadIdx.x;
  int j = (int)(i & 31);
  int t = (int)((i >> 5) & 15);
  int h = (int)((i >> 9) & 15);
  long long b = i >> 13;
  Amix[i] = x[(b * 16 + t) * 512 + h * 32 + j];
}

__device__ __forceinline__ float block_sum256(float v, float* red) {
  const int tid = threadIdx.x;
  #pragma unroll
  for (int off = 16; off; off >>= 1) v += __shfl_xor(v, off, 32);
  if ((tid & 31) == 0) red[tid >> 5] = v;
  __syncthreads();
  if (tid < 32) {
    float s = (tid < 8) ? red[tid] : 0.0f;
    #pragma unroll
    for (int off = 4; off; off >>= 1) s += __shfl_xor(s, off, 32);
    if (tid == 0) red[0] = s;
  }
  __syncthreads();
  float r = red[0];
  __syncthreads();
  return r;
}

// x1 = LN(x + P + proj_b); write to (T,B,D) layout for contiguous expert GEMMs
__global__ __launch_bounds__(256) void ln1_kernel(
    const float* __restrict__ x, const float* __restrict__ P,
    const float* __restrict__ proj_b, const float* __restrict__ gamma,
    const float* __restrict__ beta, float* __restrict__ x1t)
{
  __shared__ float red[8];
  const int row = blockIdx.x;              // b*TN + t
  const int b = row >> 4, t = row & 15;
  const int tid = threadIdx.x;
  const long long base = (long long)row * DN;
  float v0 = x[base + tid]       + P[base + tid]       + proj_b[tid];
  float v1 = x[base + 256 + tid] + P[base + 256 + tid] + proj_b[256 + tid];
  float mu  = block_sum256(v0 + v1, red) * (1.0f / DN);
  float var = block_sum256(v0 * v0 + v1 * v1, red) * (1.0f / DN) - mu * mu;
  float rs = rsqrtf(var + 1e-5f);
  const long long ob = ((long long)t * BN + b) * DN;
  x1t[ob + tid]       = (v0 - mu) * rs * gamma[tid]       + beta[tid];
  x1t[ob + 256 + tid] = (v1 - mu) * rs * gamma[256 + tid] + beta[256 + tid];
}

// gates[b,t,e] = relu(x1[b,t,:] . rW[t,:,e] + rb[t,e]); atomic aux-loss
__global__ __launch_bounds__(256) void router_kernel(
    const float* __restrict__ x1t, const float* __restrict__ rW,
    const float* __restrict__ rb, float* __restrict__ gates,
    float* __restrict__ aux)
{
  const int idx  = blockIdx.x * 8 + (threadIdx.x >> 5);  // b*TN + t
  const int lane = threadIdx.x & 31;
  const int t = idx & 15, b = idx >> 4;
  const float* xr = x1t + ((long long)t * BN + b) * DN;
  const float* w  = rW + (long long)t * DN * EN;
  float a0 = 0.f, a1 = 0.f, a2 = 0.f, a3 = 0.f;
  for (int d = lane; d < DN; d += 32) {
    float xv = xr[d];
    float4 wv = *(const float4*)(w + d * 4);
    a0 += xv * wv.x; a1 += xv * wv.y; a2 += xv * wv.z; a3 += xv * wv.w;
  }
  #pragma unroll
  for (int off = 16; off; off >>= 1) {
    a0 += __shfl_xor(a0, off, 32);
    a1 += __shfl_xor(a1, off, 32);
    a2 += __shfl_xor(a2, off, 32);
    a3 += __shfl_xor(a3, off, 32);
  }
  if (lane == 0) {
    float g0 = fmaxf(a0 + rb[t * 4 + 0], 0.0f);
    float g1 = fmaxf(a1 + rb[t * 4 + 1], 0.0f);
    float g2 = fmaxf(a2 + rb[t * 4 + 2], 0.0f);
    float g3 = fmaxf(a3 + rb[t * 4 + 3], 0.0f);
    gates[idx * 4 + 0] = g0; gates[idx * 4 + 1] = g1;
    gates[idx * 4 + 2] = g2; gates[idx * 4 + 3] = g3;
    atomicAdd(aux, (g0 + g1 + g2 + g3) * (0.01f / (float)(BN * TN)));
  }
}

// out = LN(x1 + sum_e g_e*(expert_out[t,e,b,:] + bias2[t,e,:]))
__global__ __launch_bounds__(256) void combine_ln2_kernel(
    const float* __restrict__ x1t, const float* __restrict__ eo,
    const float* __restrict__ bias2, const float* __restrict__ gates,
    const float* __restrict__ gamma, const float* __restrict__ beta,
    float* __restrict__ out)
{
  __shared__ float red[8];
  const int row = blockIdx.x;              // b*TN + t
  const int b = row >> 4, t = row & 15;
  const int tid = threadIdx.x;
  float g[4];
  #pragma unroll
  for (int e = 0; e < 4; ++e) g[e] = gates[row * 4 + e];
  float v[2];
  #pragma unroll
  for (int i = 0; i < 2; ++i) {
    int d = tid + i * 256;
    float acc = x1t[((long long)t * BN + b) * DN + d];
    #pragma unroll
    for (int e = 0; e < 4; ++e) {
      int zz = t * EN + e;
      acc += g[e] * (eo[((long long)zz * BN + b) * DN + d] +
                     bias2[(long long)zz * DN + d]);
    }
    v[i] = acc;
  }
  float mu  = block_sum256(v[0] + v[1], red) * (1.0f / DN);
  float var = block_sum256(v[0] * v[0] + v[1] * v[1], red) * (1.0f / DN) - mu * mu;
  float rs = rsqrtf(var + 1e-5f);
  #pragma unroll
  for (int i = 0; i < 2; ++i) {
    int d = tid + i * 256;
    out[(long long)row * DN + d] = (v[i] - mu) * rs * gamma[d] + beta[d];
  }
}

// =====================================================================
extern "C" void kernel_launch(void* const* d_in, const int* in_sizes, int n_in,
                              void* d_out, int out_size, void* d_ws, size_t ws_size,
                              hipStream_t stream)
{
  const float* x      = (const float*)d_in[0];
  const float* proj_W = (const float*)d_in[1];
  const float* proj_b = (const float*)d_in[2];
  const float* gamma1 = (const float*)d_in[3];
  const float* beta1  = (const float*)d_in[4];
  const float* gamma2 = (const float*)d_in[5];
  const float* beta2  = (const float*)d_in[6];
  const float* rW     = (const float*)d_in[7];
  const float* rb     = (const float*)d_in[8];
  const float* fc1    = (const float*)d_in[9];
  const float* bias1  = (const float*)d_in[10];
  const float* fc2    = (const float*)d_in[11];
  const float* bias2  = (const float*)d_in[12];

  char* ws = (char*)d_ws;
  const size_t MB = 1ull << 20;
  float*          Amix  = (float*)(ws + 0 * MB);          // 16 MB (B*T x D) f32
  float*          P     = (float*)(ws + 16 * MB);         // 16 MB proj out f32
  float*          x1t   = (float*)(ws + 32 * MB);         // 16 MB (T,B,D) f32
  float*          gates = (float*)(ws + 48 * MB);         // 128 KB (B,T,E)
  unsigned short* hid   = (unsigned short*)(ws + 49 * MB);// 128 MB (T,E,B,FH) f16
  float*          eo    = (float*)(ws + 177 * MB);        // 64 MB (T,E,B,D) f32

  float* out = (float*)d_out;
  float* aux = out + (size_t)BN * TN * DN;

  zero_aux_kernel<<<1, 1, 0, stream>>>(aux);

  // 1) head-mix permutation into GEMM A panel
  mix_kernel<<<(BN * TN * DN) / 256, 256, 0, stream>>>(x, Amix);

  // 2) projection GEMM: (8192 x 512) * (512 x 512)
  gemm_wmma<0, 0><<<dim3(DN / NBLK, (BN * TN) / MBLK, 1), 256, 0, stream>>>(
      Amix, DN, 0LL, 1, proj_W, DN, 0LL, P, DN, 0LL, nullptr, 0, DN);

  // 3) residual + LayerNorm1 -> x1 in (T,B,D)
  ln1_kernel<<<BN * TN, 256, 0, stream>>>(x, P, proj_b, gamma1, beta1, x1t);

  // 4) router gates + aux loss
  router_kernel<<<(BN * TN) / 8, 256, 0, stream>>>(x1t, rW, rb, gates, aux);

  // 5) expert fc1: 64 x [(512 x 512) * (512 x 2048)], +bias1, GELU, f16 out
  gemm_wmma<0, 1><<<dim3(FHN / NBLK, BN / MBLK, TN * EN), 256, 0, stream>>>(
      x1t, DN, (long long)BN * DN, EN,
      fc1, FHN, (long long)DN * FHN,
      hid, FHN, (long long)BN * FHN, bias1, FHN, DN);

  // 6) expert fc2: 64 x [(512 x 2048) * (2048 x 512)], A staged by TDM
  gemm_wmma<1, 0><<<dim3(DN / NBLK, BN / MBLK, TN * EN), 256, 0, stream>>>(
      hid, FHN, (long long)BN * FHN, 1,
      fc2, DN, (long long)FHN * DN,
      eo, DN, (long long)BN * DN, nullptr, 0, FHN);

  // 7) gate-weighted combine + LayerNorm2
  combine_ln2_kernel<<<BN * TN, 256, 0, stream>>>(
      x1t, eo, bias2, gates, gamma2, beta2, out);

  (void)in_sizes; (void)n_in; (void)out_size; (void)ws_size;
}
